// M_FullGaussianAttention_smilarty_11965778886677
// MI455X (gfx1250) — matrix operand order repaired and checked by reading, not compiled, not run on hardware
//
#include <hip/hip_runtime.h>

typedef __attribute__((ext_vector_type(16))) _Float16 v16h;
typedef __attribute__((ext_vector_type(8)))  _Float16 v8h;
typedef __attribute__((ext_vector_type(8)))  float    v8f;

namespace {
constexpr int Bc = 2, Lc = 2048, Hc = 8, Ec = 32, Sc = 2048, Dc = 32;
constexpr float INV_SQRT_E = 0.17677669529663689f;            // 1/sqrt(32)
constexpr float MASKED_X   = -4294967295.0f * INV_SQRT_E;     // scale * (-2^32+1)
constexpr size_t N_ROWS  = (size_t)Bc * Lc * Hc;              // 32768 rows of 32
constexpr size_t SZ_MAT  = N_ROWS * Ec * sizeof(_Float16);    // 2 MB
constexpr size_t SZ_NORM = N_ROWS * sizeof(float);            // 128 KB
constexpr size_t WS_NEED = 5 * SZ_MAT + 2 * SZ_NORM;          // Qg,Qs,Kg,Ks,Vt + q2,k2
}

// ---------------- shared helpers ----------------

__device__ __forceinline__ float halfwave_max(float v) {
  v = fmaxf(v, __shfl_xor(v, 1, 32));
  v = fmaxf(v, __shfl_xor(v, 2, 32));
  v = fmaxf(v, __shfl_xor(v, 4, 32));
  v = fmaxf(v, __shfl_xor(v, 8, 32));
  return v;
}
__device__ __forceinline__ float halfwave_sum(float v) {
  v += __shfl_xor(v, 1, 32);
  v += __shfl_xor(v, 2, 32);
  v += __shfl_xor(v, 4, 32);
  v += __shfl_xor(v, 8, 32);
  return v;
}

// A-matrix (16x32 f16) per-lane gather from an f16 row: K-chunk
// {hs*16..hs*16+15} (contiguous, matching B)
__device__ __forceinline__ v16h loadA16(const _Float16* __restrict__ rp, int hs) {
  const v8h lo = *(const v8h*)(rp + hs * 16);
  const v8h hi = *(const v8h*)(rp + hs * 16 + 8);
  return __builtin_shufflevector(lo, hi, 0, 1, 2, 3, 4, 5, 6, 7,
                                 8, 9, 10, 11, 12, 13, 14, 15);
}
// B-matrix (32x16 f16) per-lane gather: one contiguous 32B chunk at K = hs*16
__device__ __forceinline__ v16h loadB16(const _Float16* __restrict__ rp, int hs) {
  return *(const v16h*)(rp + hs * 16);
}

// ---------------- pre-pass: scale->f16, relayout, norms, V transpose ----------------

__global__ __launch_bounds__(256)
void prep_kernel(const float* __restrict__ Q, const float* __restrict__ K,
                 const float* __restrict__ V,
                 _Float16* __restrict__ Qg, _Float16* __restrict__ Qs,
                 _Float16* __restrict__ Kg, _Float16* __restrict__ Ks,
                 _Float16* __restrict__ Vt,
                 float* __restrict__ q2, float* __restrict__ k2) {
  const int idx = blockIdx.x * 256 + threadIdx.x;     // flat row in [0, B*L*H)
  const int b  = idx / (Lc * Hc);
  const int fr = idx - b * (Lc * Hc);
  const int l  = fr >> 3;          // fr / H   (== s for K/V rows)
  const int h  = fr & 7;           // fr % H
  const int src = idx * Ec;
  const int trs = (((b * Hc + h) * Lc) + l) * Ec;     // head-major position

  float qn = 0.f, kn = 0.f;
  _Float16 qr16[Ec], kr16[Ec];
#pragma unroll
  for (int i = 0; i < Ec; ++i) {
    const float qv = Q[src + i] * INV_SQRT_E;
    const float kv = K[src + i] * INV_SQRT_E;
    qn = fmaf(qv, qv, qn);
    kn = fmaf(kv, kv, kn);
    qr16[i] = (_Float16)qv;
    kr16[i] = (_Float16)kv;
  }
  q2[idx] = qn;
  k2[idx] = kn;
#pragma unroll
  for (int i = 0; i < Ec; ++i) {
    Qg[src + i] = qr16[i];   // gauss view: flat order
    Kg[src + i] = kr16[i];
    Qs[trs + i] = qr16[i];   // standard view: head-major
    Ks[trs + i] = kr16[i];
  }
#pragma unroll
  for (int d = 0; d < Dc; ++d)     // V -> [B,H,D,S] f16
    Vt[(((b * Hc + h) * Dc) + d) * Sc + l] = (_Float16)V[src + d];
}

// ---------------- main flash kernel, all-f16 operands ----------------

__global__ __launch_bounds__(32)
void fga_wmma_f16_kernel(const _Float16* __restrict__ Qs, const _Float16* __restrict__ Qg,
                         const _Float16* __restrict__ Ks, const _Float16* __restrict__ Kg,
                         const _Float16* __restrict__ Vt,
                         const float* __restrict__ q2a, const float* __restrict__ k2a,
                         float* __restrict__ O) {
  const int lane = threadIdx.x & 31;
  const int n16  = lane & 15;
  const int hs   = lane >> 4;

  const int qt = blockIdx.x & (Lc / 16 - 1);
  const int bh = blockIdx.x >> 7;
  const int h  = bh & (Hc - 1);
  const int b  = bh >> 3;
  const int l0 = qt * 16;

  __shared__ __align__(16) _Float16 Pbuf[16 * 32];

  const int headL = (b * Hc + h) * Lc;        // head-major row base (Qs/Ks)
  const int flatL = b * Lc * Hc + h * Lc;     // gauss flat row base (Qg/Kg, norms)
  const int lq = l0 + n16;

  const v16h qsA = loadA16(Qs + (headL + lq) * Ec, hs);
  const v16h qgA = loadA16(Qg + (flatL + lq) * Ec, hs);

  float q2row[8];
#pragma unroll
  for (int r = 0; r < 8; ++r) q2row[r] = q2a[flatL + l0 + r + hs * 8];

  float runmax[8], runsum[8];
  v8f o0 = {}, o1 = {};
#pragma unroll
  for (int r = 0; r < 8; ++r) { runmax[r] = -3.0e38f; runsum[r] = 0.f; }

  const v8f zacc = {};
  const int nchunks = ((l0 + 15) >> 5) + 1;   // causal: only s0 <= l0+15
  for (int c = 0; c < nchunks; ++c) {
    const int s0 = c * 32;
    const int s_a = s0 + n16;
    const int s_b = s0 + 16 + n16;

    const v16h ksB0 = loadB16(Ks + (headL + s_a) * Ec, hs);
    const v16h ksB1 = loadB16(Ks + (headL + s_b) * Ec, hs);
    const v16h kgB0 = loadB16(Kg + (flatL + s_a) * Ec, hs);
    const v16h kgB1 = loadB16(Kg + (flatL + s_b) * Ec, hs);
    const float k2t0 = k2a[flatL + s_a];      // per key column; same in both half-waves
    const float k2t1 = k2a[flatL + s_b];

    v8f s1a = __builtin_amdgcn_wmma_f32_16x16x32_f16(false, qsA, false, ksB0, (short)0, zacc, false, false);
    v8f s1b = __builtin_amdgcn_wmma_f32_16x16x32_f16(false, qsA, false, ksB1, (short)0, zacc, false, false);
    v8f cra = __builtin_amdgcn_wmma_f32_16x16x32_f16(false, qgA, false, kgB0, (short)0, zacc, false, false);
    v8f crb = __builtin_amdgcn_wmma_f32_16x16x32_f16(false, qgA, false, kgB1, (short)0, zacc, false, false);

    float pa[8], pb[8];
#pragma unroll
    for (int r = 0; r < 8; ++r) {
      const int lg = l0 + r + hs * 8;
      const float sqa = fmaxf(q2row[r] + k2t0 - 2.f * cra[r], 0.f);
      const float sqb = fmaxf(q2row[r] + k2t1 - 2.f * crb[r], 0.f);
      const float xa = (s_a > lg) ? MASKED_X : s1a[r] * __expf(-sqa) * INV_SQRT_E;
      const float xb = (s_b > lg) ? MASKED_X : s1b[r] * __expf(-sqb) * INV_SQRT_E;
      const float tm = halfwave_max(fmaxf(xa, xb));
      const float nm = fmaxf(runmax[r], tm);
      const float alpha = __expf(runmax[r] - nm);
      pa[r] = __expf(xa - nm);
      pb[r] = __expf(xb - nm);
      const float rs = halfwave_sum(pa[r] + pb[r]);
      runsum[r] = fmaf(runsum[r], alpha, rs);
      runmax[r] = nm;
      o0[r] *= alpha;
      o1[r] *= alpha;
    }

    // C-layout P -> A-layout P via 1KB LDS slab
    __syncthreads();
#pragma unroll
    for (int r = 0; r < 8; ++r) {
      const int m = r + hs * 8;
      Pbuf[m * 32 + n16]      = (_Float16)pa[r];
      Pbuf[m * 32 + 16 + n16] = (_Float16)pb[r];
    }
    __syncthreads();
    const v16h pA = loadA16(&Pbuf[n16 * 32], hs);

    // V B-tiles from transposed f16 V: contiguous 32B per lane
    const v16h vB0 = loadB16(Vt + ((size_t)(bh * Dc + n16)      * Sc + s0), hs);
    const v16h vB1 = loadB16(Vt + ((size_t)(bh * Dc + 16 + n16) * Sc + s0), hs);

    o0 = __builtin_amdgcn_wmma_f32_16x16x32_f16(false, pA, false, vB0, (short)0, o0, false, false);
    o1 = __builtin_amdgcn_wmma_f32_16x16x32_f16(false, pA, false, vB1, (short)0, o1, false, false);
  }

#pragma unroll
  for (int r = 0; r < 8; ++r) {
    const int l = l0 + r + hs * 8;
    float* op = O + (((b * Lc) + l) * Hc + h) * Dc;
    const float is = 1.f / runsum[r];
    op[n16]      = o0[r] * is;
    op[16 + n16] = o1[r] * is;
  }
}

// ---------------- fallback: round-1 direct-f32 kernel (used if ws too small) ----------------

__device__ __forceinline__ v16h load_rowA_f32(const float* __restrict__ rp, int hs, float scl) {
  v16h r;
  const float* p0 = rp + hs * 16;
  const float* p1 = rp + hs * 16 + 8;
#pragma unroll
  for (int i = 0; i < 8; ++i) r[i] = (_Float16)(p0[i] * scl);
#pragma unroll
  for (int i = 0; i < 8; ++i) r[8 + i] = (_Float16)(p1[i] * scl);
  return r;
}
__device__ __forceinline__ v16h load_rowB_f32(const float* __restrict__ rp, int hs, float scl) {
  v16h r;
  const float* p = rp + hs * 16;
#pragma unroll
  for (int i = 0; i < 16; ++i) r[i] = (_Float16)(p[i] * scl);
  return r;
}

__global__ __launch_bounds__(32)
void fga_wmma_kernel(const float* __restrict__ Q, const float* __restrict__ K,
                     const float* __restrict__ V, float* __restrict__ O) {
  const int lane = threadIdx.x & 31;
  const int n16  = lane & 15;
  const int hs   = lane >> 4;
  const int qt = blockIdx.x & (Lc / 16 - 1);
  const int bh = blockIdx.x >> 7;
  const int h  = bh & (Hc - 1);
  const int b  = bh >> 3;
  const int l0 = qt * 16;

  __shared__ __align__(16) _Float16 Pbuf[16 * 32];

  const int lq = l0 + n16;
  const v16h qsA = load_rowA_f32(Q + (((b * Lc) + lq) * Hc + h) * Ec, hs, INV_SQRT_E);
  const v16h qgA = load_rowA_f32(Q + ((b * Lc * Hc) + h * Lc + lq) * Ec, hs, INV_SQRT_E);

  float q2p = 0.f;
#pragma unroll
  for (int i = 0; i < 16; ++i) { float v = (float)qgA[i]; q2p = fmaf(v, v, q2p); }
  const float q2full = q2p + __shfl_xor(q2p, 16, 32);
  float q2row[8];
#pragma unroll
  for (int r = 0; r < 8; ++r) q2row[r] = __shfl(q2full, hs * 8 + r, 32);

  float runmax[8], runsum[8];
  v8f o0 = {}, o1 = {};
#pragma unroll
  for (int r = 0; r < 8; ++r) { runmax[r] = -3.0e38f; runsum[r] = 0.f; }

  const v8f zacc = {};
  const int nchunks = ((l0 + 15) >> 5) + 1;
  for (int c = 0; c < nchunks; ++c) {
    const int s0 = c * 32;
    const int s_a = s0 + n16;
    const int s_b = s0 + 16 + n16;
    v16h ksB0 = load_rowB_f32(K + (((b * Sc) + s_a) * Hc + h) * Ec, hs, INV_SQRT_E);
    v16h ksB1 = load_rowB_f32(K + (((b * Sc) + s_b) * Hc + h) * Ec, hs, INV_SQRT_E);
    v16h kgB0 = load_rowB_f32(K + ((b * Sc * Hc) + h * Sc + s_a) * Ec, hs, INV_SQRT_E);
    v16h kgB1 = load_rowB_f32(K + ((b * Sc * Hc) + h * Sc + s_b) * Ec, hs, INV_SQRT_E);

    float p0 = 0.f, p1 = 0.f;
#pragma unroll
    for (int i = 0; i < 16; ++i) { float v = (float)kgB0[i]; p0 = fmaf(v, v, p0); }
#pragma unroll
    for (int i = 0; i < 16; ++i) { float v = (float)kgB1[i]; p1 = fmaf(v, v, p1); }
    const float k2t0 = p0 + __shfl_xor(p0, 16, 32);
    const float k2t1 = p1 + __shfl_xor(p1, 16, 32);

    v8f s1a = __builtin_amdgcn_wmma_f32_16x16x32_f16(false, qsA, false, ksB0, (short)0, zacc, false, false);
    v8f s1b = __builtin_amdgcn_wmma_f32_16x16x32_f16(false, qsA, false, ksB1, (short)0, zacc, false, false);
    v8f cra = __builtin_amdgcn_wmma_f32_16x16x32_f16(false, qgA, false, kgB0, (short)0, zacc, false, false);
    v8f crb = __builtin_amdgcn_wmma_f32_16x16x32_f16(false, qgA, false, kgB1, (short)0, zacc, false, false);

    float pa[8], pb[8];
#pragma unroll
    for (int r = 0; r < 8; ++r) {
      const int lg = l0 + r + hs * 8;
      float sqa = fmaxf(q2row[r] + k2t0 - 2.f * cra[r], 0.f);
      float sqb = fmaxf(q2row[r] + k2t1 - 2.f * crb[r], 0.f);
      float xa = (s_a > lg) ? MASKED_X : s1a[r] * __expf(-sqa) * INV_SQRT_E;
      float xb = (s_b > lg) ? MASKED_X : s1b[r] * __expf(-sqb) * INV_SQRT_E;
      float tm = halfwave_max(fmaxf(xa, xb));
      float nm = fmaxf(runmax[r], tm);
      float alpha = __expf(runmax[r] - nm);
      pa[r] = __expf(xa - nm);
      pb[r] = __expf(xb - nm);
      float rs = halfwave_sum(pa[r] + pb[r]);
      runsum[r] = fmaf(runsum[r], alpha, rs);
      runmax[r] = nm;
      o0[r] *= alpha;
      o1[r] *= alpha;
    }

    __syncthreads();
#pragma unroll
    for (int r = 0; r < 8; ++r) {
      const int m = r + hs * 8;
      Pbuf[m * 32 + n16]      = (_Float16)pa[r];
      Pbuf[m * 32 + 16 + n16] = (_Float16)pb[r];
    }
    __syncthreads();
    const v16h pA = loadA16(&Pbuf[n16 * 32], hs);

    v16h vB0, vB1;
#pragma unroll
    for (int i = 0; i < 16; ++i) {
      const int s = s0 + hs * 16 + i;
      const float* vp = V + (((b * Sc) + s) * Hc + h) * Dc;
      vB0[i] = (_Float16)vp[n16];
      vB1[i] = (_Float16)vp[16 + n16];
    }
    o0 = __builtin_amdgcn_wmma_f32_16x16x32_f16(false, pA, false, vB0, (short)0, o0, false, false);
    o1 = __builtin_amdgcn_wmma_f32_16x16x32_f16(false, pA, false, vB1, (short)0, o1, false, false);
  }

#pragma unroll
  for (int r = 0; r < 8; ++r) {
    const int l = l0 + r + hs * 8;
    float* op = O + (((b * Lc) + l) * Hc + h) * Dc;
    const float is = 1.f / runsum[r];
    op[n16]      = o0[r] * is;
    op[16 + n16] = o1[r] * is;
  }
}

// ---------------- launcher ----------------

extern "C" void kernel_launch(void* const* d_in, const int* in_sizes, int n_in,
                              void* d_out, int out_size, void* d_ws, size_t ws_size,
                              hipStream_t stream) {
  (void)in_sizes; (void)n_in; (void)out_size;
  const float* Q = (const float*)d_in[0];
  const float* K = (const float*)d_in[1];
  const float* V = (const float*)d_in[2];
  float*       O = (float*)d_out;

  const dim3 grid(Bc * Hc * (Lc / 16));   // 2048 waves, one 16-query tile each
  const dim3 block(32);                   // one wave32 per workgroup

  if (d_ws != nullptr && ws_size >= WS_NEED) {
    char* w = (char*)d_ws;
    _Float16* Qg = (_Float16*)(w);
    _Float16* Qs = (_Float16*)(w + 1 * SZ_MAT);
    _Float16* Kg = (_Float16*)(w + 2 * SZ_MAT);
    _Float16* Ks = (_Float16*)(w + 3 * SZ_MAT);
    _Float16* Vt = (_Float16*)(w + 4 * SZ_MAT);
    float*    q2 = (float*)   (w + 5 * SZ_MAT);
    float*    k2 = (float*)   (w + 5 * SZ_MAT + SZ_NORM);

    prep_kernel<<<dim3((unsigned)(N_ROWS / 256)), dim3(256), 0, stream>>>(
        Q, K, V, Qg, Qs, Kg, Ks, Vt, q2, k2);
    fga_wmma_f16_kernel<<<grid, block, 0, stream>>>(Qs, Qg, Ks, Kg, Vt, q2, k2, O);
  } else {
    fga_wmma_kernel<<<grid, block, 0, stream>>>(Q, K, V, O);
  }
}
